// ScaledDotProductAttention_20942260536094
// MI455X (gfx1250) — matrix operand order repaired and checked by reading, not compiled, not run on hardware
//
#include <hip/hip_runtime.h>

typedef __attribute__((ext_vector_type(16))) _Float16 v16h;
typedef __attribute__((ext_vector_type(8)))  _Float16 v8h;
typedef __attribute__((ext_vector_type(8)))  float    v8f;

#define NB   32      // batch
#define NH   16      // heads
#define NN   512     // sequence
#define ND   64      // head dim
#define QT   32      // queries per workgroup
#define KC   128     // keys per LDS chunk
#define KROW 72      // padded Keff row (halves)
#define VROW 136     // padded VeffT row (halves)
#define SROW 520     // padded score row (floats / halves)

__launch_bounds__(256)
__global__ void attn_pos_kernel(const float* __restrict__ Q,
                                const float* __restrict__ K,
                                const float* __restrict__ V,
                                const int*   __restrict__ pos_ids,
                                const float* __restrict__ pos_k_table,
                                const float* __restrict__ pos_v_table,
                                float* __restrict__ ctx_out,
                                float* __restrict__ attn_out)
{
    __shared__ _Float16 Qs[QT * KROW];      //  4.5 KB  Q tile (f16, pre-scaled)
    __shared__ _Float16 KVs[KC * KROW];     // 18.0 KB  Keff chunk OR VeffT chunk
    __shared__ float    Ss[QT * SROW];      // 65.0 KB  score block
    __shared__ _Float16 Ph[QT * SROW];      // 32.5 KB  softmax probs (f16)
    __shared__ float    red[256];
    __shared__ float    rinv[QT];

    const int tid  = threadIdx.x;
    const int wave = tid >> 5;
    const int lane = tid & 31;

    const int bh = blockIdx.x >> 4;         // (b*16+h), 512 total
    const int qt = blockIdx.x & 15;
    const int h  = bh & 15;
    const int q0 = qt * QT;

    const long head_base = (long)bh * NN * ND;
    const long qtile_base = head_base + (long)q0 * ND;

    // ---- Q tile -> LDS f16, scale 1/sqrt(64) folded in ----
    for (int i = tid; i < QT * ND; i += 256) {
        int r = i >> 6, d = i & 63;
        Qs[r * KROW + d] = (_Float16)(Q[qtile_base + i] * 0.125f);
    }
    __syncthreads();

    // per-wave GEMM1 assignment: 8 waves = 2 q-rows x (key tiles mod 4)
    const int qrow  = wave & 1;             // which 16-query block
    const int ktsel = wave >> 1;            // key tile within chunk: ktsel, ktsel+4

    // ---- A fragments from Q tile (constant for whole kernel) ----
    v16h a0, a1;
    {
        const int r  = lane & 15;
        const int kb = (lane >> 4) * 8;
        const int row = (qrow * 16 + r) * KROW;
        v8h x0 = *(const v8h*)&Qs[row + kb];
        v8h x1 = *(const v8h*)&Qs[row + 16 + kb];
        v8h x2 = *(const v8h*)&Qs[row + 32 + kb];
        v8h x3 = *(const v8h*)&Qs[row + 48 + kb];
#pragma unroll
        for (int i = 0; i < 8; ++i) {
            a0[i] = x0[i]; a0[8 + i] = x1[i];
            a1[i] = x2[i]; a1[8 + i] = x3[i];
        }
    }

    // ================= GEMM1: S = (Q/8) * Keff^T =================
    for (int kc = 0; kc < 4; ++kc) {
        if (kc) __syncthreads();            // protect KVs reuse
        // build Keff chunk (K + pos_k_table[pos_ids]) in f16, row-major
        for (int i = tid; i < KC * ND; i += 256) {
            int kr = i >> 6, d = i & 63;
            int key = kc * KC + kr;
            int pid = pos_ids[h * NN + key];
            float kv = K[head_base + (long)key * ND + d] + pos_k_table[pid * ND + d];
            KVs[kr * KROW + d] = (_Float16)kv;
        }
        __syncthreads();

#pragma unroll
        for (int t = 0; t < 2; ++t) {
            const int kt   = ktsel + 4 * t;       // key tile (16 keys) in chunk
            const int coln = lane & 15;
            const int koff = (lane >> 4) * 8;
            const _Float16* kp = &KVs[(kt * 16 + coln) * KROW + koff];
            v8h y0 = *(const v8h*)(kp);
            v8h y1 = *(const v8h*)(kp + 16);
            v8h y2 = *(const v8h*)(kp + 32);
            v8h y3 = *(const v8h*)(kp + 48);
            v16h b0, b1;
#pragma unroll
            for (int i = 0; i < 8; ++i) {
                b0[i] = y0[i]; b0[8 + i] = y1[i];
                b1[i] = y2[i]; b1[8 + i] = y3[i];
            }
            v8f c = {};
            c = __builtin_amdgcn_wmma_f32_16x16x32_f16(false, a0, false, b0, (short)0, c, false, false);
            c = __builtin_amdgcn_wmma_f32_16x16x32_f16(false, a1, false, b1, (short)0, c, false, false);
            const int srow0 = qrow * 16 + ((lane >> 4) << 3);
            const int scol  = kc * KC + kt * 16 + coln;
#pragma unroll
            for (int i = 0; i < 8; ++i)
                Ss[(srow0 + i) * SROW + scol] = c[i];
        }
    }
    __syncthreads();

    // ================= softmax over 512-key rows =================
    const int srow  = tid >> 3;             // 32 rows, 8 threads each
    const int spart = tid & 7;
    float* srp = &Ss[srow * SROW];

    float m = -3.4e38f;
#pragma unroll 8
    for (int j = 0; j < 64; ++j) m = fmaxf(m, srp[spart + 8 * j]);
    red[tid] = m;
    __syncthreads();
    float rm = red[srow * 8];
#pragma unroll
    for (int j = 1; j < 8; ++j) rm = fmaxf(rm, red[srow * 8 + j]);
    __syncthreads();

    float s = 0.0f;
#pragma unroll 8
    for (int j = 0; j < 64; ++j) {
        int idx = spart + 8 * j;
        float e = __expf(srp[idx] - rm);
        srp[idx] = e;
        s += e;
    }
    red[tid] = s;
    __syncthreads();
    if (spart == 0) {
        float tsum = 0.0f;
#pragma unroll
        for (int j = 0; j < 8; ++j) tsum += red[srow * 8 + j];
        rinv[srow] = 1.0f / tsum;
    }
    __syncthreads();

    // coalesced attn write + f16 P for GEMM2
    const long attn_base = ((long)bh * NN + q0) * NN;
    for (int i = tid; i < QT * NN; i += 256) {
        int r = i >> 9, c = i & 511;
        float p = Ss[r * SROW + c] * rinv[r];
        attn_out[attn_base + i] = p;
        Ph[r * SROW + c] = (_Float16)p;
    }
    __syncthreads();

    // ================= GEMM2: context = P * Veff =================
    const int dcol = wave >> 1;             // 4 d-columns x 2 q-rows = 8 tiles
    v8f acc = {};
    for (int vc = 0; vc < 4; ++vc) {
        if (vc) __syncthreads();
        // build VeffT chunk: VeffT[d][k] = V[key][d] + pos_v_table[pid][d]
        for (int i = tid; i < KC * ND; i += 256) {
            int kr = i >> 6, d = i & 63;
            int key = vc * KC + kr;
            int pid = pos_ids[h * NN + key];
            float vv = V[head_base + (long)key * ND + d] + pos_v_table[pid * ND + d];
            KVs[d * VROW + kr] = (_Float16)vv;
        }
        __syncthreads();

#pragma unroll
        for (int ks = 0; ks < 4; ++ks) {
            const int k0 = ks * 32;
            // A fragment from Ph
            const int r  = lane & 15;
            const int kb = (lane >> 4) * 8;
            const int prow = (qrow * 16 + r) * SROW + vc * KC + k0;
            v8h x0 = *(const v8h*)&Ph[prow + kb];
            v8h x1 = *(const v8h*)&Ph[prow + 16 + kb];
            v16h a;
#pragma unroll
            for (int i = 0; i < 8; ++i) { a[i] = x0[i]; a[8 + i] = x1[i]; }
            // B fragment from VeffT
            const int coln = lane & 15;
            const int koff = (lane >> 4) * 8;
            const _Float16* vt = &KVs[(dcol * 16 + coln) * VROW + k0 + koff];
            v8h y0 = *(const v8h*)(vt);
            v8h y1 = *(const v8h*)(vt + 16);
            v16h bf;
#pragma unroll
            for (int i = 0; i < 8; ++i) { bf[i] = y0[i]; bf[8 + i] = y1[i]; }
            acc = __builtin_amdgcn_wmma_f32_16x16x32_f16(false, a, false, bf, (short)0, acc, false, false);
        }
    }

    // write context tile
    {
        const int coln = lane & 15;
        const int mofs = (lane >> 4) * 8;
        const long cbase = ((long)bh * NN + q0 + qrow * 16) * ND + dcol * 16 + coln;
#pragma unroll
        for (int i = 0; i < 8; ++i)
            ctx_out[cbase + (long)(i + mofs) * ND] = acc[i];
    }
}

extern "C" void kernel_launch(void* const* d_in, const int* in_sizes, int n_in,
                              void* d_out, int out_size, void* d_ws, size_t ws_size,
                              hipStream_t stream) {
    const float* Q       = (const float*)d_in[0];
    const float* K       = (const float*)d_in[1];
    const float* V       = (const float*)d_in[2];
    const int*   pos_ids = (const int*)d_in[3];
    const float* pk      = (const float*)d_in[4];
    const float* pv      = (const float*)d_in[5];

    float* ctx  = (float*)d_out;
    float* attn = ctx + (size_t)NB * NH * NN * ND;   // context first, then attn

    dim3 grid(NB * NH * (NN / QT));   // 8192 workgroups
    dim3 block(256);                  // 8 waves (wave32)
    hipLaunchKernelGGL(attn_pos_kernel, grid, block, 0, stream,
                       Q, K, V, pos_ids, pk, pv, ctx, attn);
}
